// GPT_8151847928208
// MI455X (gfx1250) — matrix operand order repaired and checked by reading, not compiled, not run on hardware
//
#include <hip/hip_runtime.h>
#include <hip/hip_bf16.h>
#include <math.h>

// ---------------------------------------------------------------------------
// Model constants (match reference)
// ---------------------------------------------------------------------------
constexpr int B_    = 2;
constexpr int T_    = 1024;
constexpr int D_    = 1024;
constexpr int L_    = 6;
constexpr int NH_   = 16;
constexpr int NKV_  = 8;
constexpr int HD_   = 64;
constexpr int KV_   = NKV_ * HD_;      // 512
constexpr int RECUR_= 2;
constexpr int NVB_  = 5;
constexpr int V_    = 32768;
constexpr int MLP_  = 3 * D_;          // 3072
constexpr int TOK_  = B_ * T_;         // 2048
constexpr int BGK_  = 4096;            // BG_BK
constexpr int BGD_  = 128;             // BG_DIM
constexpr float EPS_ = 1.1920929e-07f;
constexpr float SOFTCAP_ = 30.0f;
constexpr int VCHUNK_ = 4096;          // vocab chunk for logits

// ---------------------------------------------------------------------------
// WMMA types / helpers (CDNA5, wave32)
// ---------------------------------------------------------------------------
typedef __attribute__((ext_vector_type(16))) __bf16 v16bf;
typedef __attribute__((ext_vector_type(8)))  float  v8f;

__device__ __forceinline__ v8f wmma_bf16(v16bf a, v16bf b, v8f c) {
    // emits v_wmma_f32_16x16x32_bf16
    return __builtin_amdgcn_wmma_f32_16x16x32_bf16(
        /*neg_a=*/false, a, /*neg_b=*/false, b,
        /*c_mod=*/(short)0, c, /*reuse_a=*/false, /*reuse_b=*/false);
}

// 16-bit A-matrix 16x32 per-lane K mapping (contiguous, matches B fills):
// lanes 0-15 hold K = {0..15}; lanes 16-31 hold K = {16..31}.
__device__ __forceinline__ int amap(int e, int half) {
    return e + half * 16;
}

// ---------------------------------------------------------------------------
// Block reductions (blockDim.x == 256, 8 wave32)
// ---------------------------------------------------------------------------
__device__ __forceinline__ float block_sum256(float v, float* red) {
    #pragma unroll
    for (int o = 16; o > 0; o >>= 1) v += __shfl_xor(v, o);
    int wave = threadIdx.x >> 5, lane = threadIdx.x & 31;
    if (lane == 0) red[wave] = v;
    __syncthreads();
    float tot = 0.f;
    #pragma unroll
    for (int i = 0; i < 8; ++i) tot += red[i];
    __syncthreads();
    return tot;
}

__device__ __forceinline__ float block_max256(float v, float* red) {
    #pragma unroll
    for (int o = 16; o > 0; o >>= 1) v = fmaxf(v, __shfl_xor(v, o));
    int wave = threadIdx.x >> 5, lane = threadIdx.x & 31;
    if (lane == 0) red[wave] = v;
    __syncthreads();
    float tot = -1e30f;
    #pragma unroll
    for (int i = 0; i < 8; ++i) tot = fmaxf(tot, red[i]);
    __syncthreads();
    return tot;
}

// ---------------------------------------------------------------------------
// Generic tiled WMMA GEMM: C[M,N] = A[M,K] @ B[K,N], fp32 in/out, bf16x3 core.
// BT=false: B[k,n] at B[k*ldb + n] (row-major weight).
// BT=true : B[k,n] at B[n*ldb + k] (e.g. emb^T for the vocab head).
// Block = 256 threads = 8 waves; wave w owns rows [m0+16w, +16), cols 64.
// Requires M%128==0, N%64==0, K%32==0 (all shapes here satisfy this).
// ---------------------------------------------------------------------------
template <bool BT>
__global__ __launch_bounds__(256)
void gemm_kernel(const float* __restrict__ A, const float* __restrict__ Bm,
                 float* __restrict__ C, int K, int lda, int ldb, int ldc) {
    const int wave = threadIdx.x >> 5;
    const int lane = threadIdx.x & 31;
    const int mrow = lane & 15;
    const int half = lane >> 4;
    const int m0 = blockIdx.y * 128 + wave * 16;
    const int n0 = blockIdx.x * 64;

    v8f acc[4] = {};

    const float* arow = A + (size_t)(m0 + mrow) * lda;

    for (int kk = 0; kk < K; kk += 32) {
        v16bf ahi, alo;
        #pragma unroll
        for (int e = 0; e < 16; ++e) {
            const float av = arow[kk + amap(e, half)];
            const __bf16 h = (__bf16)av;
            ahi[e] = h;
            alo[e] = (__bf16)(av - (float)h);
        }
        #pragma unroll
        for (int s = 0; s < 4; ++s) {
            const int n = n0 + s * 16 + mrow;
            v16bf bhi, blo;
            #pragma unroll
            for (int e = 0; e < 16; ++e) {
                const int kidx = kk + half * 16 + e;
                const float bv = BT ? Bm[(size_t)n * ldb + kidx]
                                    : Bm[(size_t)kidx * ldb + n];
                const __bf16 h = (__bf16)bv;
                bhi[e] = h;
                blo[e] = (__bf16)(bv - (float)h);
            }
            acc[s] = wmma_bf16(ahi, bhi, acc[s]);
            acc[s] = wmma_bf16(ahi, blo, acc[s]);
            acc[s] = wmma_bf16(alo, bhi, acc[s]);
        }
    }

    #pragma unroll
    for (int s = 0; s < 4; ++s) {
        const int n = n0 + s * 16 + mrow;
        #pragma unroll
        for (int i = 0; i < 8; ++i) {
            const int row = m0 + i + 8 * half;   // C layout: M = i + 8*(lane/16)
            C[(size_t)row * ldc + n] = acc[s][i];
        }
    }
}

// ---------------------------------------------------------------------------
// Embedding + hashed-bigram embed + RMS.  One block (256 thr) per token.
// x  = rms(emb[id] + bg_scale * (bg_embed[h] @ bg_proj));  x0 = x
// ---------------------------------------------------------------------------
__global__ __launch_bounds__(256)
void embed_kernel(const int* __restrict__ ids, const float* __restrict__ emb,
                  const float* __restrict__ bg_embed, const float* __restrict__ bg_proj,
                  const float* __restrict__ bg_scale,
                  float* __restrict__ x, float* __restrict__ x0) {
    __shared__ float bgrow[BGD_];
    __shared__ float red[8];
    const int tok = blockIdx.x;
    const int t = tok % T_;
    const int id = ids[tok];
    int hidx = 0;
    if (t > 0) {
        const int prev = ids[tok - 1];
        hidx = ((id * 36313) ^ (prev * 27191)) % (BGK_ - 1);
    }
    if (threadIdx.x < BGD_) bgrow[threadIdx.x] = bg_embed[(size_t)hidx * BGD_ + threadIdx.x];
    __syncthreads();

    const float scale = bg_scale[0];
    float vals[4];
    float ss = 0.f;
    #pragma unroll
    for (int j = 0; j < 4; ++j) {
        const int d = threadIdx.x + j * 256;
        float acc = 0.f;
        for (int kx = 0; kx < BGD_; ++kx)
            acc = fmaf(bgrow[kx], bg_proj[(size_t)kx * D_ + d], acc);
        float xv = emb[(size_t)id * D_ + d] + scale * acc;
        vals[j] = xv;
        ss += xv * xv;
    }
    const float tot = block_sum256(ss, red);
    const float r = rsqrtf(tot / (float)D_ + EPS_);
    #pragma unroll
    for (int j = 0; j < 4; ++j) {
        const int d = threadIdx.x + j * 256;
        const float o = vals[j] * r;
        x[(size_t)tok * D_ + d] = o;
        x0[(size_t)tok * D_ + d] = o;
    }
}

// x = m0*x + m1*x0 (in place) ; xn = rms(x).  One block per token.
__global__ __launch_bounds__(256)
void mix_rms_kernel(float* __restrict__ x, const float* __restrict__ x0,
                    const float* __restrict__ mixl, float* __restrict__ xn) {
    __shared__ float red[8];
    const int tok = blockIdx.x;
    float vals[4];
    float ss = 0.f;
    #pragma unroll
    for (int j = 0; j < 4; ++j) {
        const int d = threadIdx.x + j * 256;
        const float v = mixl[d] * x[(size_t)tok * D_ + d] +
                        mixl[D_ + d] * x0[(size_t)tok * D_ + d];
        vals[j] = v;
        ss += v * v;
    }
    const float tot = block_sum256(ss, red);
    const float r = rsqrtf(tot / (float)D_ + EPS_);
    #pragma unroll
    for (int j = 0; j < 4; ++j) {
        const int d = threadIdx.x + j * 256;
        x[(size_t)tok * D_ + d] = vals[j];
        xn[(size_t)tok * D_ + d] = vals[j] * r;
    }
}

// xn = rms(x). One block per token.
__global__ __launch_bounds__(256)
void rms_kernel(const float* __restrict__ x, float* __restrict__ xn) {
    __shared__ float red[8];
    const int tok = blockIdx.x;
    float vals[4];
    float ss = 0.f;
    #pragma unroll
    for (int j = 0; j < 4; ++j) {
        const int d = threadIdx.x + j * 256;
        vals[j] = x[(size_t)tok * D_ + d];
        ss += vals[j] * vals[j];
    }
    const float tot = block_sum256(ss, red);
    const float r = rsqrtf(tot / (float)D_ + EPS_);
    #pragma unroll
    for (int j = 0; j < 4; ++j) {
        const int d = threadIdx.x + j * 256;
        xn[(size_t)tok * D_ + d] = vals[j] * r;
    }
}

// v[tok,:] += ve_l[ids[tok], :]   (KV_ = 512). One block per token.
__global__ __launch_bounds__(256)
void vel_add_kernel(float* __restrict__ v, const float* __restrict__ ve_l,
                    const int* __restrict__ ids) {
    const int tok = blockIdx.x;
    const int id = ids[tok];
    for (int j = threadIdx.x; j < KV_; j += 256)
        v[(size_t)tok * KV_ + j] += ve_l[(size_t)id * KV_ + j];
}

// Per-head RMS + RoPE (+ optional head gain). One wave per (token, head).
// buf layout: [b, t, nheads, HD_].  g==nullptr for K.
__global__ __launch_bounds__(256)
void rope_kernel(float* __restrict__ buf, int nheads,
                 const float* __restrict__ g, float gmul, int nvec) {
    const int wave = threadIdx.x >> 5;
    const int lane = threadIdx.x & 31;
    const int vid = blockIdx.x * 8 + wave;
    if (vid >= nvec) return;
    const int h = vid % nheads;
    const int t = (vid / nheads) % T_;
    float* base = buf + (size_t)vid * HD_;

    float x1 = base[lane];
    float x2 = base[lane + 32];
    float ss = x1 * x1 + x2 * x2;
    #pragma unroll
    for (int o = 16; o > 0; o >>= 1) ss += __shfl_xor(ss, o);
    const float r = rsqrtf(ss / (float)HD_ + EPS_);
    x1 *= r; x2 *= r;

    // inv[j] = 10000^{-j/32}, ang = t * inv[j]
    const float inv = __expf(-((float)lane / 32.f) * 9.210340371976184f);
    const float ang = (float)t * inv;
    const float c = __cosf(ang), s = __sinf(ang);
    const float gg = g ? g[h] * gmul : 1.0f;

    base[lane]      = ( x1 * c + x2 * s) * gg;
    base[lane + 32] = (-x1 * s + x2 * c) * gg;
}

// ---------------------------------------------------------------------------
// Flash attention with WMMA.  One wave per (b, h, 16-query tile).
// q: [b,t,NH,HD] (pre-scaled by qg/sqrt(HD)), k,v: [b,t,NKV,HD], y: [b,t,NH,HD]
// Computes S^T = K_tile @ Q_tile^T per 32-kv chunk, online softmax per query
// (query == lane column), Y^T accumulated as V^T @ P^T.
// ---------------------------------------------------------------------------
__global__ __launch_bounds__(256)
void attn_kernel(const float* __restrict__ q, const float* __restrict__ k,
                 const float* __restrict__ v, float* __restrict__ y) {
    const int wave = threadIdx.x >> 5;
    const int lane = threadIdx.x & 31;
    const int mrow = lane & 15;
    const int half = lane >> 4;

    const int tile = blockIdx.x * 8 + wave;   // 0 .. B*NH*(T/16)-1
    const int qt = tile % (T_ / 16);
    const int h  = (tile / (T_ / 16)) % NH_;
    const int b  = tile / ((T_ / 16) * NH_);
    const int kvh = h / (NH_ / NKV_);
    const int q0 = qt * 16;
    const int qglob = q0 + mrow;              // this lane's query column

    // Q as B-fragments (two 32-d chunks): col = mrow, k = half*16 + e
    v16bf qb[2];
    #pragma unroll
    for (int c = 0; c < 2; ++c) {
        const float* qp = q + (((size_t)(b * T_ + q0 + mrow) * NH_ + h) * HD_) + 32 * c + half * 16;
        #pragma unroll
        for (int e = 0; e < 16; ++e) qb[c][e] = (__bf16)qp[e];
    }

    v8f o[4] = {};
    float m_run = -1e30f, s_run = 0.f;

    const int kend = q0 + 16;                 // exclusive causal bound
    for (int kb = 0; kb < kend; kb += 32) {
        // scores S^T for kv rows [kb, kb+16) and [kb+16, kb+32)
        v8f st0 = {}, st1 = {};
        #pragma unroll
        for (int c = 0; c < 2; ++c) {
            v16bf a0, a1;
            const float* k0 = k + (((size_t)(b * T_ + kb + mrow) * NKV_ + kvh) * HD_) + 32 * c;
            const float* k1 = k0 + (size_t)16 * NKV_ * HD_;
            #pragma unroll
            for (int e = 0; e < 16; ++e) {
                const int kd = amap(e, half);
                a0[e] = (__bf16)k0[kd];
                a1[e] = (__bf16)k1[kd];
            }
            st0 = wmma_bf16(a0, qb[c], st0);
            st1 = wmma_bf16(a1, qb[c], st1);
        }

        // causal mask: kv row of acc element i is kb + i + 8*half (+16 for st1)
        float cm = -1e30f;
        #pragma unroll
        for (int i = 0; i < 8; ++i) {
            const int kg0 = kb + i + 8 * half;
            if (kg0 > qglob) st0[i] = -1e30f;
            if (kg0 + 16 > qglob) st1[i] = -1e30f;
            cm = fmaxf(cm, fmaxf(st0[i], st1[i]));
        }
        cm = fmaxf(cm, __shfl_xor(cm, 16));   // per-query chunk max
        const float nm = fmaxf(m_run, cm);
        const float rescale = __expf(m_run - nm);

        float p0[8], p1[8];
        float cs = 0.f;
        #pragma unroll
        for (int i = 0; i < 8; ++i) {
            p0[i] = __expf(st0[i] - nm);
            p1[i] = __expf(st1[i] - nm);
            cs += p0[i] + p1[i];
        }
        cs += __shfl_xor(cs, 16);             // per-query chunk sum
        s_run = s_run * rescale + cs;
        m_run = nm;
        #pragma unroll
        for (int dt = 0; dt < 4; ++dt)
            #pragma unroll
            for (int i = 0; i < 8; ++i) o[dt][i] *= rescale;

        // assemble P^T as a 32x16 B fragment (col = mrow, k = half*16 + e)
        v16bf pb;
        #pragma unroll
        for (int e = 0; e < 8; ++e) {
            const float oth0 = __shfl_xor(p0[e], 16);
            const float oth1 = __shfl_xor(p1[e], 16);
            const float lo_v = (half == 0) ? p0[e] : oth1;  // rows e of its tile
            const float hi_v = (half == 0) ? oth0 : p1[e];  // rows e+8
            pb[e]     = (__bf16)lo_v;
            pb[e + 8] = (__bf16)hi_v;
        }

        // Y^T += V^T @ P^T  (A rows = d, contraction over the 32 kv)
        #pragma unroll
        for (int dt = 0; dt < 4; ++dt) {
            v16bf av;
            const int d = dt * 16 + mrow;
            #pragma unroll
            for (int e = 0; e < 16; ++e) {
                const int kvl = amap(e, half);
                av[e] = (__bf16)v[(((size_t)(b * T_ + kb + kvl) * NKV_ + kvh) * HD_) + d];
            }
            o[dt] = wmma_bf16(av, pb, o[dt]);
        }
    }

    const float inv_s = 1.0f / s_run;
    #pragma unroll
    for (int dt = 0; dt < 4; ++dt) {
        #pragma unroll
        for (int i = 0; i < 8; ++i) {
            const int d = dt * 16 + i + 8 * half;
            y[(((size_t)(b * T_ + qglob) * NH_ + h) * HD_) + d] = o[dt][i] * inv_s;
        }
    }
}

// x[i] += s[i % D] * t[i]
__global__ __launch_bounds__(256)
void residual_add_kernel(float* __restrict__ x, const float* __restrict__ t,
                         const float* __restrict__ s, int n) {
    const int i = blockIdx.x * 256 + threadIdx.x;
    if (i < n) x[i] += s[i % D_] * t[i];
}

// h = relu(h)^2
__global__ __launch_bounds__(256)
void relu2_kernel(float* __restrict__ hbuf, int n) {
    const int i = blockIdx.x * 256 + threadIdx.x;
    if (i < n) {
        const float h = hbuf[i];
        hbuf[i] = h > 0.f ? h * h : 0.f;
    }
}

// stats[tok] = {running max, running sumexp, z_target}
__global__ __launch_bounds__(256)
void init_stats_kernel(float* __restrict__ stats) {
    const int tok = blockIdx.x * 256 + threadIdx.x;
    if (tok < TOK_) {
        stats[tok * 3 + 0] = -1e30f;
        stats[tok * 3 + 1] = 0.f;
        stats[tok * 3 + 2] = 0.f;
    }
}

// Softcap + streaming logsumexp over one vocab chunk. One block per token.
__global__ __launch_bounds__(256)
void softcap_stats_kernel(const float* __restrict__ big, float* __restrict__ stats,
                          const int* __restrict__ tgt, int chunk_start) {
    __shared__ float sm[VCHUNK_];
    __shared__ float red[8];
    __shared__ float stgt;
    const int tok = blockIdx.x;
    if (threadIdx.x == 0) stgt = -2e30f;
    __syncthreads();

    const float* row = big + (size_t)tok * VCHUNK_;
    const int tg = tgt[tok];
    for (int i = threadIdx.x; i < VCHUNK_; i += 256) {
        const float z = SOFTCAP_ * tanhf(row[i] * (1.f / SOFTCAP_));
        sm[i] = z;
        if (chunk_start + i == tg) stgt = z;   // at most one writer
    }
    __syncthreads();

    float mx = -1e30f;
    for (int i = threadIdx.x; i < VCHUNK_; i += 256) mx = fmaxf(mx, sm[i]);
    const float bm = block_max256(mx, red);

    float se = 0.f;
    for (int i = threadIdx.x; i < VCHUNK_; i += 256) se += __expf(sm[i] - bm);
    const float bs = block_sum256(se, red);

    if (threadIdx.x == 0) {
        const float m_run = stats[tok * 3 + 0];
        const float s_run = stats[tok * 3 + 1];
        const float nm = fmaxf(m_run, bm);
        stats[tok * 3 + 1] = s_run * __expf(m_run - nm) + bs * __expf(bm - nm);
        stats[tok * 3 + 0] = nm;
        if (stgt > -1.5e30f) stats[tok * 3 + 2] = stgt;
    }
}

__global__ __launch_bounds__(256)
void nll_kernel(const float* __restrict__ stats, float* __restrict__ nll) {
    const int tok = blockIdx.x * 256 + threadIdx.x;
    if (tok < TOK_) {
        const float m = stats[tok * 3 + 0];
        const float s = stats[tok * 3 + 1];
        const float zt = stats[tok * 3 + 2];
        nll[tok] = (m + logf(s)) - zt;       // -log p(target)
    }
}

__global__ __launch_bounds__(256)
void mean_kernel(const float* __restrict__ nll, float* __restrict__ out) {
    __shared__ float red[256];
    float a = 0.f;
    for (int i = threadIdx.x; i < TOK_; i += 256) a += nll[i];
    red[threadIdx.x] = a;
    __syncthreads();
    for (int s = 128; s > 0; s >>= 1) {
        if ((int)threadIdx.x < s) red[threadIdx.x] += red[threadIdx.x + s];
        __syncthreads();
    }
    if (threadIdx.x == 0) out[0] = red[0] / (float)TOK_;
}

// ---------------------------------------------------------------------------
// Host-side launch
// ---------------------------------------------------------------------------
static inline void launch_gemm(const float* A, const float* Bm, float* C,
                               int M, int N, int K, int lda, int ldb, int ldc,
                               bool bt, hipStream_t s) {
    dim3 grid(N / 64, M / 128);
    if (bt) gemm_kernel<true ><<<grid, 256, 0, s>>>(A, Bm, C, K, lda, ldb, ldc);
    else    gemm_kernel<false><<<grid, 256, 0, s>>>(A, Bm, C, K, lda, ldb, ldc);
}

extern "C" void kernel_launch(void* const* d_in, const int* in_sizes, int n_in,
                              void* d_out, int out_size, void* d_ws, size_t ws_size,
                              hipStream_t stream) {
    const int*   ids      = (const int*)  d_in[0];
    const int*   tgt      = (const int*)  d_in[1];
    const float* emb      = (const float*)d_in[2];
    const float* bg_embed = (const float*)d_in[3];
    const float* bg_proj  = (const float*)d_in[4];
    const float* bg_scale = (const float*)d_in[5];
    const float* ve       = (const float*)d_in[6];
    const float* Wq       = (const float*)d_in[7];
    const float* Wk       = (const float*)d_in[8];
    const float* Wv       = (const float*)d_in[9];
    const float* Wo       = (const float*)d_in[10];
    const float* qg       = (const float*)d_in[11];
    const float* a_s      = (const float*)d_in[12];
    const float* m_s      = (const float*)d_in[13];
    const float* mix      = (const float*)d_in[14];
    const float* Wfc      = (const float*)d_in[15];
    const float* Wpr      = (const float*)d_in[16];
    float* out = (float*)d_out;
    float* ws  = (float*)d_ws;

    // workspace layout (floats); total ~21.0M floats (~84 MB)
    const size_t TD = (size_t)TOK_ * D_;        // 2,097,152
    float* x    = ws;                           // TD
    float* x0   = x    + TD;                    // TD
    float* xn   = x0   + TD;                    // TD (also o-proj / mlp-out tmp)
    float* qb   = xn   + TD;                    // TD
    float* kb   = qb   + TD;                    // TOK_*KV_
    float* vb   = kb   + (size_t)TOK_ * KV_;    // TOK_*KV_
    float* yb   = vb   + (size_t)TOK_ * KV_;    // TD
    float* big  = yb   + TD;                    // TOK_*4096 (mlp hidden / logits chunk)
    float* stats= big  + (size_t)TOK_ * VCHUNK_;// TOK_*3
    float* nll  = stats+ (size_t)TOK_ * 3;      // TOK_

    const float qscale = 0.125f;                // 1/sqrt(HD)

    // 1) embeddings + bigram + rms
    embed_kernel<<<TOK_, 256, 0, stream>>>(ids, emb, bg_embed, bg_proj, bg_scale, x, x0);

    // 2) layers (RECUR * L effective steps)
    for (int eff = 0; eff < RECUR_ * L_; ++eff) {
        const int l = eff % L_;
        const float* Wq_l  = Wq  + (size_t)l * D_ * D_;
        const float* Wk_l  = Wk  + (size_t)l * D_ * KV_;
        const float* Wv_l  = Wv  + (size_t)l * D_ * KV_;
        const float* Wo_l  = Wo  + (size_t)l * D_ * D_;
        const float* Wfc_l = Wfc + (size_t)l * D_ * MLP_;
        const float* Wpr_l = Wpr + (size_t)l * MLP_ * D_;
        const float* mix_l = mix + (size_t)l * 2 * D_;
        const float* qg_l  = qg  + (size_t)l * NH_;
        const float* as_l  = a_s + (size_t)l * D_;
        const float* ms_l  = m_s + (size_t)l * D_;
        const float* ve_l  = ve  + (size_t)(eff % NVB_) * V_ * KV_;

        mix_rms_kernel<<<TOK_, 256, 0, stream>>>(x, x0, mix_l, xn);

        launch_gemm(xn, Wq_l, qb, TOK_, D_,  D_, D_, D_,  D_,  false, stream);
        launch_gemm(xn, Wk_l, kb, TOK_, KV_, D_, D_, KV_, KV_, false, stream);
        launch_gemm(xn, Wv_l, vb, TOK_, KV_, D_, D_, KV_, KV_, false, stream);

        vel_add_kernel<<<TOK_, 256, 0, stream>>>(vb, ve_l, ids);

        rope_kernel<<<(TOK_ * NH_) / 8, 256, 0, stream>>>(qb, NH_,  qg_l, qscale, TOK_ * NH_);
        rope_kernel<<<(TOK_ * NKV_) / 8, 256, 0, stream>>>(kb, NKV_, nullptr, 1.0f, TOK_ * NKV_);

        attn_kernel<<<(B_ * NH_ * (T_ / 16)) / 8, 256, 0, stream>>>(qb, kb, vb, yb);

        launch_gemm(yb, Wo_l, xn, TOK_, D_, D_, D_, D_, D_, false, stream);
        residual_add_kernel<<<(int)(TD / 256), 256, 0, stream>>>(x, xn, as_l, (int)TD);

        rms_kernel<<<TOK_, 256, 0, stream>>>(x, xn);
        launch_gemm(xn, Wfc_l, big, TOK_, MLP_, D_, D_, MLP_, MLP_, false, stream);
        relu2_kernel<<<(TOK_ * MLP_) / 256, 256, 0, stream>>>(big, TOK_ * MLP_);
        launch_gemm(big, Wpr_l, xn, TOK_, D_, MLP_, MLP_, D_, D_, false, stream);
        residual_add_kernel<<<(int)(TD / 256), 256, 0, stream>>>(x, xn, ms_l, (int)TD);
    }

    // 3) vocab head: rms -> chunked logits (x @ emb^T) -> softcap -> logsumexp
    rms_kernel<<<TOK_, 256, 0, stream>>>(x, xn);
    init_stats_kernel<<<(TOK_ + 255) / 256, 256, 0, stream>>>(stats);
    for (int ch = 0; ch < V_ / VCHUNK_; ++ch) {
        const float* embT = emb + (size_t)ch * VCHUNK_ * D_;   // B[k,n] = embT[n*D + k]
        launch_gemm(xn, embT, big, TOK_, VCHUNK_, D_, D_, D_, VCHUNK_, true, stream);
        softcap_stats_kernel<<<TOK_, 256, 0, stream>>>(big, stats, tgt, ch * VCHUNK_);
    }
    nll_kernel<<<(TOK_ + 255) / 256, 256, 0, stream>>>(stats, nll);
    mean_kernel<<<1, 256, 0, stream>>>(nll, out);
}
